// SpanFusionLM_88510686036287
// MI455X (gfx1250) — matrix operand order repaired and checked by reading, not compiled, not run on hardware
//
#include <hip/hip_runtime.h>
#include <hip/hip_bf16.h>
#include <math.h>

#define B_        2
#define NCTX      1024
#define KSPAN     32
#define HDIM      2048
#define VOCAB     50257
#define VPAD      50272          // pad N stride to 32
#define GSTEPS    8
#define ROWS      64             // B_ * KSPAN
#define PRED_TOK  50256
#define TOPP      0.9f

#define KC        256            // A-panel chunk (columns) for k_gemm_et
#define NCHUNK    (HDIM / KC)    // 8
#define KCBYTES   (64 * KC * 2)  // 32 KB per buffer

#define KC2       128            // A-panel chunk for k_gemm_w_tanh (dual panel)
#define NCHUNK2   (HDIM / KC2)   // 16
#define PANB2     (64 * KC2 * 2) // 16 KB per panel

typedef __bf16 bf16;
typedef __attribute__((ext_vector_type(16))) __bf16 v16bf;
typedef __attribute__((ext_vector_type(8)))  __bf16 v8bf;
typedef __attribute__((ext_vector_type(8)))  float  v8f;
typedef __attribute__((ext_vector_type(4)))  unsigned int v4u;
typedef __attribute__((ext_vector_type(4)))  int v4i;
typedef __attribute__((ext_vector_type(8)))  int v8i;

// ---------------------------------------------------------------- helpers

// A-matrix (16x32 bf16) fragment per ISA layout:
// lanes 0-15: K chunks {0..7, 16..23}; lanes 16-31: K chunks {8..15, 24..31}
__device__ __forceinline__ v16bf mk_afrag(const bf16* row, int k, int hsel) {
  v8bf lo = *(const v8bf*)(row + k + hsel * 8);
  v8bf hi = *(const v8bf*)(row + k + 16 + hsel * 8);
  v16bf r;
#pragma unroll
  for (int i = 0; i < 8; ++i) { r[i] = lo[i]; r[i + 8] = hi[i]; }
  return r;
}

// B-matrix (32x16 bf16) fragment: lane = column; same K order as A:
// lanes 0-15 hold K {0..7, 16..23}; lanes 16-31 hold K {8..15, 24..31}.
__device__ __forceinline__ v16bf mk_bfrag(const bf16* __restrict__ row, int k, int hsel) {
  v8bf lo = *(const v8bf*)(row + k + hsel * 8);
  v8bf hi = *(const v8bf*)(row + k + 16 + hsel * 8);
  v16bf r;
#pragma unroll
  for (int i = 0; i < 8; ++i) { r[i] = lo[i]; r[i + 8] = hi[i]; }
  return r;
}

// Issue a TDM 2D tile load: 64 rows x kc bf16 (row stride HDIM bf16) into LDS
// at byte offset lds_off.  Descriptor per CDNA5 ISA (D# groups 0/1).
__device__ __forceinline__ void tdm_load_panel(const bf16* A, int chunk, int kc,
                                               unsigned lds_off) {
  unsigned long long ga = (unsigned long long)(const void*)(A + (size_t)chunk * kc);
  const unsigned td0   = (unsigned)kc / 2;   // dwords per tile row (data_size = 4B)
  const unsigned td1   = 64;                 // rows
  const unsigned tile0 = (unsigned)kc / 2;
  const unsigned tile1 = 64;
  const unsigned str0  = HDIM / 2;           // row stride in dwords
  v4u g0 = { 1u,                                        // count=1, user descriptor
             lds_off,                                   // lds_addr
             (unsigned)ga,                              // global_addr[31:0]
             ((unsigned)(ga >> 32) & 0x01FFFFFFu) | (2u << 30) };  // [56:32] | type=2
  v8i g1 = { (int)(2u << 16),                           // wg_mask=0, data_size=4B
             (int)(td0 << 16),                          // tensor_dim0[15:0]
             (int)((td0 >> 16) | (td1 << 16)),          // tensor_dim0[31:16] | dim1
             (int)((td1 >> 16) | (tile0 << 16)),        // tensor_dim1[31:16] | tile0
             (int)(tile1),                              // tile1 | tile2=0
             (int)(str0),                               // dim0 stride lo
             0, 0 };                                    // stride hi / dim1 stride
  v4i z4 = { 0, 0, 0, 0 };
#if __clang_major__ >= 23
  v8i z8 = { 0, 0, 0, 0, 0, 0, 0, 0 };
  __builtin_amdgcn_tensor_load_to_lds(g0, g1, z4, z4, z8, 0);
#else
  __builtin_amdgcn_tensor_load_to_lds(g0, g1, z4, z4, 0);
#endif
}

// Escape an LDS buffer + full compiler memory barrier: forces the compiler to
// treat the TDM engine's writes as visible stores so staged-panel loads are
// not folded to undef.
#define LDS_ESCAPE(ptr)  { void* _p = (void*)(ptr); asm volatile("" : : "v"(_p) : "memory"); }
#define MEM_BARRIER()    asm volatile("" ::: "memory")

__device__ __forceinline__ unsigned pcg_hash(unsigned x) {
  x = x * 747796405u + 2891336453u;
  unsigned w = ((x >> ((x >> 28u) + 4u)) ^ x) * 277803737u;
  return (w >> 22u) ^ w;
}

__device__ __forceinline__ float gumbel_of(int t, int r, int v) {
  unsigned h = pcg_hash((unsigned)v * 2654435761u ^ (unsigned)r * 40503u ^
                        (unsigned)(t + 1) * 668265263u);
  float u = (float)(h >> 8) * (1.0f / 16777216.0f);
  u = fmaxf(u, 1e-12f);
  return -logf(-logf(u));
}

// ---------------------------------------------------------------- setup kernels

__global__ void k_conv_e(const float* __restrict__ E, bf16* __restrict__ E16, long n) {
  long i = ((long)blockIdx.x * blockDim.x + threadIdx.x) * 4;
  if (i + 3 < n) {
    float4 f = *(const float4*)(E + i);
    E16[i]     = (bf16)f.x;
    E16[i + 1] = (bf16)f.y;
    E16[i + 2] = (bf16)f.z;
    E16[i + 3] = (bf16)f.w;
  } else {
    for (long j = i; j < n; ++j) E16[j] = (bf16)E[j];
  }
}

// WT[n][k] = (bf16) W[k][n]   (so WMMA B-operand columns are contiguous rows)
__global__ void k_transpose_w(const float* __restrict__ W, bf16* __restrict__ WT) {
  __shared__ float tile[32][33];
  int bx = blockIdx.x * 32, by = blockIdx.y * 32;
  int tx = threadIdx.x, ty = threadIdx.y;            // 32 x 8
  for (int i = 0; i < 32; i += 8)
    tile[ty + i][tx] = W[(size_t)(by + ty + i) * HDIM + bx + tx];
  __syncthreads();
  for (int i = 0; i < 32; i += 8)
    WT[(size_t)(bx + ty + i) * HDIM + by + tx] = (bf16)tile[tx][ty + i];
}

// S_prompt[b][h] = sum over the 1024 prompt tokens of E[tok][h]
__global__ void k_prefix(const int* __restrict__ seqP, const float* __restrict__ E,
                         float* __restrict__ Sp) {
  int b = blockIdx.x / 8;
  int h = (blockIdx.x % 8) * 256 + threadIdx.x;
  float acc = 0.f;
  for (int t = 0; t < NCTX; ++t) {
    int tok = seqP[b * NCTX + t];
    acc += E[(size_t)tok * HDIM + h];
  }
  Sp[b * HDIM + h] = acc;
}

__global__ void k_init(const float* __restrict__ E, int* __restrict__ seqPred,
                       int* __restrict__ filled, bf16* __restrict__ z16,
                       float* __restrict__ zF) {
  int r = blockIdx.x;
  if (threadIdx.x == 0) { seqPred[r] = PRED_TOK; filled[r] = 0; }
  for (int c = 0; c < 8; ++c) {
    int h = threadIdx.x + c * 256;
    float v = E[(size_t)PRED_TOK * HDIM + h];
    z16[(size_t)r * HDIM + h] = (bf16)v;
    zF[(size_t)r * HDIM + h] = v;
  }
}

// h16[r][:] = bf16( (S_prompt + prefix of predicted-token embeddings) / (n+j+1) )
__global__ void k_hmean(const int* __restrict__ seqPred, const float* __restrict__ E,
                        const float* __restrict__ Sp, bf16* __restrict__ h16) {
  int r = blockIdx.x, b = r / KSPAN, j = r % KSPAN;
  for (int c = 0; c < 8; ++c) {
    int h = threadIdx.x + c * 256;
    float acc = Sp[b * HDIM + h];
    for (int i = 0; i <= j; ++i) {
      int tok = seqPred[b * KSPAN + i];
      acc += E[(size_t)tok * HDIM + h];
    }
    h16[(size_t)r * HDIM + h] = (bf16)(acc / (float)(NCTX + j + 1));
  }
}

// ---------------------------------------------------------------- WMMA GEMMs

// D16[64][H] (+ optional DF fp32) = tanh( A1 @ B1t.T  (+ A2 @ B2t.T) )
// Both A panels are staged in LDS by the TDM (double-buffered); the weight
// stream (B) is register-pipelined one step ahead.
__global__ void k_gemm_w_tanh(const bf16* __restrict__ A1, const bf16* __restrict__ B1t,
                              const bf16* __restrict__ A2, const bf16* __restrict__ B2t,
                              bf16* __restrict__ D16, float* __restrict__ DF) {
  __shared__ __align__(128) bf16 sP[2][2][64 * KC2];   // [buf][panel] = 64 KB
  int wid = threadIdx.x >> 5, lane = threadIdx.x & 31;
  int nt = blockIdx.x * 8 + wid;
  int col0 = nt * 16;
  int csel = lane & 15, hsel = lane >> 4;

  const bf16* b1row = B1t + (size_t)(col0 + csel) * HDIM;
  const bf16* b2row = A2 ? (B2t + (size_t)(col0 + csel) * HDIM) : nullptr;

  v8f acc[4];
#pragma unroll
  for (int m = 0; m < 4; ++m)
#pragma unroll
    for (int i = 0; i < 8; ++i) acc[m][i] = 0.f;

  LDS_ESCAPE(sP);
  if (threadIdx.x == 0) {
    tdm_load_panel(A1, 0, KC2, 0u);
    if (A2) tdm_load_panel(A2, 0, KC2, (unsigned)PANB2);
  }

  v16bf b1cur = mk_bfrag(b1row, 0, hsel);
  v16bf b2cur = A2 ? mk_bfrag(b2row, 0, hsel) : b1cur;

  for (int c = 0; c < NCHUNK2; ++c) {
    if (threadIdx.x == 0) __builtin_amdgcn_s_wait_tensorcnt(0);
    __syncthreads();
    MEM_BARRIER();
    if (threadIdx.x == 0 && c + 1 < NCHUNK2) {
      unsigned boff = (unsigned)(((c + 1) & 1) * 2 * PANB2);
      tdm_load_panel(A1, c + 1, KC2, boff);
      if (A2) tdm_load_panel(A2, c + 1, KC2, boff + (unsigned)PANB2);
    }
    const bf16* s1 = sP[c & 1][0];
    const bf16* s2 = sP[c & 1][1];
    for (int kk = 0; kk < KC2; kk += 32) {
      int k = c * KC2 + kk;
      int kn = (k + 32 < HDIM) ? k + 32 : k;
      v16bf b1next = mk_bfrag(b1row, kn, hsel);
      v16bf b2next = A2 ? mk_bfrag(b2row, kn, hsel) : b1next;
#pragma unroll
      for (int m = 0; m < 4; ++m) {
        v16bf a1 = mk_afrag(s1 + (size_t)(m * 16 + csel) * KC2, kk, hsel);
        acc[m] = __builtin_amdgcn_wmma_f32_16x16x32_bf16(
            false, a1, false, b1cur, (short)0, acc[m], false, false);
      }
      if (A2) {
#pragma unroll
        for (int m = 0; m < 4; ++m) {
          v16bf a2 = mk_afrag(s2 + (size_t)(m * 16 + csel) * KC2, kk, hsel);
          acc[m] = __builtin_amdgcn_wmma_f32_16x16x32_bf16(
              false, a2, false, b2cur, (short)0, acc[m], false, false);
        }
      }
      b1cur = b1next;
      b2cur = b2next;
    }
  }

#pragma unroll
  for (int m = 0; m < 4; ++m)
#pragma unroll
    for (int r = 0; r < 8; ++r) {
      int row = m * 16 + r + hsel * 8;
      int col = col0 + csel;
      float v = tanhf(acc[m][r]);
      D16[(size_t)row * HDIM + col] = (bf16)v;
      if (DF) DF[(size_t)row * HDIM + col] = v;
    }
}

// OUT[64][VPAD] fp32 = A[64][H] @ E16.T
// TDM double-buffers the 64xKC A-panel in LDS; the E stream (bandwidth driver)
// is software-pipelined so loads stay in flight across the WMMA chain.
__global__ void k_gemm_et(const bf16* __restrict__ A, const bf16* __restrict__ E16,
                          float* __restrict__ OUT) {
  __shared__ __align__(128) bf16 sA[2][64 * KC];     // 2 x 32 KB
  int wid = threadIdx.x >> 5, lane = threadIdx.x & 31;
  int nt = blockIdx.x * 8 + wid;
  int col0 = nt * 16;
  int csel = lane & 15, hsel = lane >> 4;
  int col = col0 + csel;
  int vc = col < VOCAB ? col : (VOCAB - 1);          // clamp loads; mask stores
  const bf16* brow = E16 + (size_t)vc * HDIM;

  v8f acc[4];
#pragma unroll
  for (int m = 0; m < 4; ++m)
#pragma unroll
    for (int i = 0; i < 8; ++i) acc[m][i] = 0.f;

  LDS_ESCAPE(sA);
  if (threadIdx.x == 0) tdm_load_panel(A, 0, KC, 0u);   // TDM chunk 0 -> buf 0

  v16bf bcur = mk_bfrag(brow, 0, hsel);              // pipeline head for E stream

  for (int c = 0; c < NCHUNK; ++c) {
    if (threadIdx.x == 0) __builtin_amdgcn_s_wait_tensorcnt(0);
    __syncthreads();                                  // buf[c&1] ready; buf[(c+1)&1] free
    MEM_BARRIER();
    if (threadIdx.x == 0 && c + 1 < NCHUNK)
      tdm_load_panel(A, c + 1, KC, (unsigned)(((c + 1) & 1) * KCBYTES));

    const bf16* sbuf = sA[c & 1];
    for (int kk = 0; kk < KC; kk += 32) {
      int k = c * KC + kk;
      int kn = (k + 32 < HDIM) ? k + 32 : k;
      v16bf bnext = mk_bfrag(brow, kn, hsel);         // prefetch next E fragment
#pragma unroll
      for (int m = 0; m < 4; ++m) {
        const bf16* arow = sbuf + (size_t)(m * 16 + csel) * KC;
        v16bf a = mk_afrag(arow, kk, hsel);           // ds_load from staged panel
        acc[m] = __builtin_amdgcn_wmma_f32_16x16x32_bf16(
            false, a, false, bcur, (short)0, acc[m], false, false);
      }
      bcur = bnext;
    }
  }

  if (col < VOCAB) {
#pragma unroll
    for (int m = 0; m < 4; ++m)
#pragma unroll
      for (int r = 0; r < 8; ++r) {
        int row = m * 16 + r + hsel * 8;
        OUT[(size_t)row * VPAD + col] = acc[m][r];
      }
  }
}

// ---------------------------------------------------------------- reductions

__global__ void k_entropy(const float* __restrict__ L, const int* __restrict__ filled,
                          float* __restrict__ ent, int update) {
  __shared__ float sd[256];
  int r = blockIdx.x, tid = threadIdx.x;
  const float* row = L + (size_t)r * VPAD;

  float mx = -1e30f;
  for (int v = tid; v < VOCAB; v += 256) mx = fmaxf(mx, row[v]);
  sd[tid] = mx; __syncthreads();
  for (int o = 128; o; o >>= 1) { if (tid < o) sd[tid] = fmaxf(sd[tid], sd[tid + o]); __syncthreads(); }
  mx = sd[0]; __syncthreads();

  float s1 = 0.f, s2 = 0.f;
  for (int v = tid; v < VOCAB; v += 256) {
    float d = row[v] - mx;
    float e = __expf(d);
    s1 += e; s2 += d * e;
  }
  sd[tid] = s1; __syncthreads();
  for (int o = 128; o; o >>= 1) { if (tid < o) sd[tid] += sd[tid + o]; __syncthreads(); }
  float S = sd[0]; __syncthreads();
  sd[tid] = s2; __syncthreads();
  for (int o = 128; o; o >>= 1) { if (tid < o) sd[tid] += sd[tid + o]; __syncthreads(); }
  if (tid == 0) {
    float h = logf(S) - sd[0] / S;
    if (!update || !filled[r]) ent[r] = h;
  }
}

__global__ void k_pick(const float* __restrict__ ent, const int* __restrict__ filled,
                       int* __restrict__ pick, int m) {
  int tid = threadIdx.x;
  if (tid < ROWS) pick[tid] = 0;
  __syncthreads();
  if (tid < B_) {
    float e[KSPAN];
    for (int j = 0; j < KSPAN; ++j)
      e[j] = filled[tid * KSPAN + j] ? 1e30f : ent[tid * KSPAN + j];
    for (int it = 0; it < m; ++it) {
      int am = 0;
      float bv = e[0];
      for (int j = 1; j < KSPAN; ++j) if (e[j] < bv) { bv = e[j]; am = j; }
      pick[tid * KSPAN + am] = 1;
      e[am] = 1e30f;
    }
  }
}

__global__ void k_sample(const float* __restrict__ L, const int* __restrict__ pick,
                         int* __restrict__ seqPred, int* __restrict__ filled, int t) {
  __shared__ float sv[256];
  __shared__ int   si[256];
  int r = blockIdx.x, tid = threadIdx.x;
  const float* row = L + (size_t)r * VPAD;

  float mx = -1e30f;
  for (int v = tid; v < VOCAB; v += 256) mx = fmaxf(mx, row[v]);
  sv[tid] = mx; __syncthreads();
  for (int o = 128; o; o >>= 1) { if (tid < o) sv[tid] = fmaxf(sv[tid], sv[tid + o]); __syncthreads(); }
  mx = sv[0]; __syncthreads();

  float s1 = 0.f;
  for (int v = tid; v < VOCAB; v += 256) s1 += __expf(row[v] - mx);
  sv[tid] = s1; __syncthreads();
  for (int o = 128; o; o >>= 1) { if (tid < o) sv[tid] += sv[tid + o]; __syncthreads(); }
  float S = sv[0]; __syncthreads();

  float lo = mx - 30.f, hi = mx;
  for (int it = 0; it < 24; ++it) {
    float mid = 0.5f * (lo + hi);
    float part = 0.f;
    for (int v = tid; v < VOCAB; v += 256) {
      float l = row[v];
      if (l >= mid) part += __expf(l - mx);
    }
    sv[tid] = part; __syncthreads();
    for (int o = 128; o; o >>= 1) { if (tid < o) sv[tid] += sv[tid + o]; __syncthreads(); }
    float mass = sv[0]; __syncthreads();
    if (mass > TOPP * S) lo = mid; else hi = mid;
  }
  float tau = lo;

  float bv = -1e30f; int bi = 0;
  for (int v = tid; v < VOCAB; v += 256) {
    float l = row[v];
    if (l >= tau) {
      float g = l + gumbel_of(t, r, v);
      if (g > bv) { bv = g; bi = v; }
    }
  }
  sv[tid] = bv; si[tid] = bi; __syncthreads();
  for (int o = 128; o; o >>= 1) {
    if (tid < o && sv[tid + o] > sv[tid]) { sv[tid] = sv[tid + o]; si[tid] = si[tid + o]; }
    __syncthreads();
  }
  if (tid == 0 && pick[r]) { seqPred[r] = si[0]; filled[r] = 1; }
}

// ---------------------------------------------------------------- output

__global__ void k_writeout(const int* __restrict__ seqPrompt, const int* __restrict__ seqPred,
                           const float* __restrict__ L, const float* __restrict__ zF,
                           float* __restrict__ out) {
  long idx = (long)blockIdx.x * 256 + threadIdx.x;
  const long n_seq = (long)B_ * (NCTX + KSPAN);
  const long n_log = (long)ROWS * VOCAB;
  const long n_z   = (long)ROWS * HDIM;
  if (idx < n_seq) {
    int b = (int)(idx / (NCTX + KSPAN)), p = (int)(idx % (NCTX + KSPAN));
    int v = (p < NCTX) ? seqPrompt[b * NCTX + p] : seqPred[b * KSPAN + (p - NCTX)];
    out[idx] = (float)v;
  } else if (idx < n_seq + n_log) {
    long q = idx - n_seq;
    int r = (int)(q / VOCAB), v = (int)(q % VOCAB);
    out[idx] = L[(size_t)r * VPAD + v];
  } else if (idx < n_seq + n_log + n_z) {
    out[idx] = zF[idx - n_seq - n_log];
  }
}

// ---------------------------------------------------------------- host

extern "C" void kernel_launch(void* const* d_in, const int* in_sizes, int n_in,
                              void* d_out, int out_size, void* d_ws, size_t ws_size,
                              hipStream_t stream) {
  (void)in_sizes; (void)n_in; (void)out_size; (void)ws_size;

  const int*   seqP = (const int*)d_in[0];
  const float* E    = (const float*)d_in[1];
  const float* Wd   = (const float*)d_in[2];
  const float* Wz   = (const float*)d_in[3];
  const float* Wh   = (const float*)d_in[4];
  float* out = (float*)d_out;

  char* ws = (char*)d_ws;
  size_t off = 0;
  auto give = [&](size_t bytes) -> void* {
    void* p = ws + off;
    off += (bytes + 255) & ~(size_t)255;
    return p;
  };

  bf16*  E16    = (bf16*)give((size_t)VOCAB * HDIM * 2);
  bf16*  WdT    = (bf16*)give((size_t)HDIM * HDIM * 2);
  bf16*  WzT    = (bf16*)give((size_t)HDIM * HDIM * 2);
  bf16*  WhT    = (bf16*)give((size_t)HDIM * HDIM * 2);
  float* Sp     = (float*)give((size_t)B_ * HDIM * 4);
  bf16*  h16    = (bf16*)give((size_t)ROWS * HDIM * 2);
  bf16*  hdec16 = (bf16*)give((size_t)ROWS * HDIM * 2);
  bf16*  hdec0  = (bf16*)give((size_t)ROWS * HDIM * 2);
  bf16*  z16a   = (bf16*)give((size_t)ROWS * HDIM * 2);
  bf16*  z16b   = (bf16*)give((size_t)ROWS * HDIM * 2);
  float* zF     = (float*)give((size_t)ROWS * HDIM * 4);
  float* logits = (float*)give((size_t)ROWS * VPAD * 4);
  float* ent    = (float*)give(ROWS * 4);
  int*   sPred  = (int*)give(ROWS * 4);
  int*   fill   = (int*)give(ROWS * 4);
  int*   pick   = (int*)give(ROWS * 4);

  const long VH = (long)VOCAB * HDIM;
  k_conv_e<<<(int)((VH / 4 + 255) / 256), 256, 0, stream>>>(E, E16, VH);
  dim3 tb(32, 8), tg(64, 64);
  k_transpose_w<<<tg, tb, 0, stream>>>(Wd, WdT);
  k_transpose_w<<<tg, tb, 0, stream>>>(Wz, WzT);
  k_transpose_w<<<tg, tb, 0, stream>>>(Wh, WhT);
  k_prefix<<<B_ * 8, 256, 0, stream>>>(seqP, E, Sp);
  k_init<<<ROWS, 256, 0, stream>>>(E, sPred, fill, z16a, zF);

  const int GB_W  = HDIM / 128;               // 16 blocks, 8 waves x 16 cols
  const int GB_ET = (VOCAB + 127) / 128;      // 393 blocks

  // Initial decode + entropy (h_dec_for_enc = hdec0 is frozen afterwards)
  k_hmean<<<ROWS, 256, 0, stream>>>(sPred, E, Sp, h16);
  k_gemm_w_tanh<<<GB_W, 256, 0, stream>>>(h16, WdT, nullptr, nullptr, hdec0, nullptr);
  k_gemm_et<<<GB_ET, 256, 0, stream>>>(hdec0, E16, logits);
  k_entropy<<<ROWS, 256, 0, stream>>>(logits, fill, ent, 0);

  bf16* zin = z16a;
  bf16* zout = z16b;
  for (int t = 0; t < GSTEPS; ++t) {
    k_pick<<<1, 64, 0, stream>>>(ent, fill, pick, 4);  // m_t = 4 for K=32, g=8
    k_gemm_w_tanh<<<GB_W, 256, 0, stream>>>(zin, WzT, hdec0, WhT, zout, zF);
    k_gemm_et<<<GB_ET, 256, 0, stream>>>(zout, E16, logits);
    k_sample<<<ROWS, 256, 0, stream>>>(logits, pick, sPred, fill, t);
    if (t < GSTEPS - 1) {
      k_hmean<<<ROWS, 256, 0, stream>>>(sPred, E, Sp, h16);
      k_gemm_w_tanh<<<GB_W, 256, 0, stream>>>(h16, WdT, nullptr, nullptr, hdec16, nullptr);
      k_gemm_et<<<GB_ET, 256, 0, stream>>>(hdec16, E16, logits);
      k_entropy<<<ROWS, 256, 0, stream>>>(logits, fill, ent, 1);
    }
    bf16* tmp = zin; zin = zout; zout = tmp;
  }
  // After the final iteration `logits` holds z @ E.T == logits_span (T=1).

  const long total = (long)B_ * (NCTX + KSPAN) + (long)ROWS * VOCAB + (long)ROWS * HDIM;
  k_writeout<<<(int)((total + 255) / 256), 256, 0, stream>>>(seqP, sPred, logits, zF, out);
}
